// MaxCosineSimilarityBlock_2826088480904
// MI455X (gfx1250) — matrix operand order repaired and checked by reading, not compiled, not run on hardware
//
#include <hip/hip_runtime.h>
#include <hip/hip_bf16.h>

typedef __attribute__((ext_vector_type(16))) _Float16     v16h;
typedef __attribute__((ext_vector_type(8)))  float        v8f;
typedef __attribute__((ext_vector_type(4)))  unsigned int v4u;
typedef __attribute__((ext_vector_type(8)))  int          v8i;
typedef __attribute__((ext_vector_type(4)))  int          v4i;

#define B_   32
#define C_   16
#define L_   4096
#define S_   64
#define K_   64
#define W_   4033          // L - S + 1
#define WTILES 253         // ceil(W/16)
#define EPS_ 1e-8f

#define WAVES_PB 4         // waves per block in the WMMA kernel
#define TILES_PW 2         // w-tiles per wave
#define TILES_PB (WAVES_PB * TILES_PW)
#define SHN_BYTES (C_ * K_ * S_ * 2)   // 131072

// ---- TDM availability probes (device pass only; host pass falls back) ----
#if defined(__has_builtin)
#  if __has_builtin(__builtin_amdgcn_tensor_load_to_lds) && \
      __has_builtin(__builtin_amdgcn_s_wait_tensorcnt)
#    define HAVE_TDM 1
#  endif
#endif
#ifndef HAVE_TDM
#  define HAVE_TDM 0
#endif
#if defined(__has_include)
#  if __has_include(<hip/amd_detail/amd_gfx1250_TDM.h>)
#    define TDM_6ARG 1
#  endif
#endif
#ifndef TDM_6ARG
#  define TDM_6ARG 0
#endif

// Diagnostic: report (device pass only, via stderr) which staging path lowered.
#ifdef __AMDGCN__
#  if HAVE_TDM
#    if TDM_6ARG
#      warning "CDNA5: TDM path enabled (6-arg tensor_load_to_lds)"
#    else
#      warning "CDNA5: TDM path enabled (5-arg tensor_load_to_lds)"
#    endif
#  else
#    warning "CDNA5: TDM builtin NOT available; using manual LDS staging"
#  endif
#endif

// ---------------------------------------------------------------- kernel 0
__global__ void zero_out_kernel(float* __restrict__ out) {
    int i = blockIdx.x * blockDim.x + threadIdx.x;
    if (i < B_ * K_) out[i] = 0.0f;
}

// ---------------------------------------------------------------- kernel 1
// Normalize shapelets along S, store as f16 [C][K][S].
__global__ void shapelet_norm_kernel(const float* __restrict__ sh,
                                     _Float16* __restrict__ shn) {
    int idx = blockIdx.x * blockDim.x + threadIdx.x;   // c*K + k
    if (idx >= C_ * K_) return;
    const float* r = sh + (size_t)idx * S_;
    float s = 0.0f;
#pragma unroll 8
    for (int i = 0; i < S_; ++i) s = fmaf(r[i], r[i], s);
    float invn = 1.0f / fmaxf(sqrtf(s), EPS_);
#pragma unroll 8
    for (int i = 0; i < S_; ++i) shn[(size_t)idx * S_ + i] = (_Float16)(r[i] * invn);
}

// ---------------------------------------------------------------- kernel 2
// Per-(b,c) sliding-window inverse L2 norms via LDS staging.
__global__ void window_invnorm_kernel(const float* __restrict__ x,
                                      float* __restrict__ inv) {
    __shared__ float lds[256 + 64];
    const int bc = blockIdx.y;
    const int w0 = blockIdx.x * 256;
    const int t  = threadIdx.x;
    const float* xr = x + (size_t)bc * L_;

    int g = w0 + t;
    lds[t] = (g < L_) ? xr[g] : 0.0f;
    if (t < 64) {
        int g2 = w0 + 256 + t;
        lds[256 + t] = (g2 < L_) ? xr[g2] : 0.0f;
    }
    __syncthreads();

    int w = w0 + t;
    if (w < W_) {
        float s = 0.0f;
#pragma unroll 8
        for (int i = 0; i < S_; ++i) {
            float v = lds[t + i];
            s = fmaf(v, v, s);
        }
        inv[(size_t)bc * W_ + w] = 1.0f / fmaxf(sqrtf(s), EPS_);
    }
}

// ---------------------------------------------------------------- kernel 3
// Block = 4 waves sharing an LDS copy of shn (TDM-staged when available).
// Each wave computes TWO 16x64 sim tiles. Per channel c: preload all 8 B
// fragments from LDS in one clause, then run 16 v_wmma_f32_16x16x32_f16
// back-to-back while A-pack VALU co-executes. relu/(1/C) + row-reduce +
// atomic-max epilogue.
__global__ void __launch_bounds__(WAVES_PB * 32)
sim_wmma_kernel(const float* __restrict__ x,
                const _Float16* __restrict__ shn,
                const float* __restrict__ inv,
                float* __restrict__ out) {
    extern __shared__ __align__(16) _Float16 lshn[];   // C*K*S f16 = 128 KB

    const int b      = blockIdx.y;
    const int wv     = threadIdx.x >> 5;
    const int lane   = threadIdx.x & 31;
    const int halfId = lane >> 4;
    const int r16    = lane & 15;

    // ---- stage normalized shapelets into LDS ----
#if HAVE_TDM
    if (threadIdx.x == 0) {
        unsigned long long ga = (unsigned long long)(const void*)shn;
        v4u g0;
        g0[0] = 1u;                                   // count=1, user descriptor
        g0[1] = (unsigned int)(uintptr_t)(void*)lshn; // LDS byte address
        g0[2] = (unsigned int)(ga & 0xFFFFFFFFull);   // global_addr[31:0]
        g0[3] = (unsigned int)((ga >> 32) & 0x1FFFFFFull) | (2u << 30); // type=2
        v8i g1;
        g1[0] = 0x00010000;        // data_size=1 (2 bytes), no mask/flags
        g1[1] = 0x10000000;        // tensor_dim0 = 4096 (low 16 in [31:16])
        g1[2] = 0x00100000;        // tensor_dim1 = 16   (low 16 in [31:16])
        g1[3] = 0x10000000;        // tile_dim0 = 4096
        g1[4] = 16;                // tile_dim1 = 16
        g1[5] = 4096;              // tensor_dim0_stride = 4096
        g1[6] = 0;
        g1[7] = 0;
        v4i z4 = {0, 0, 0, 0};
#if TDM_6ARG
        v8i z8 = {0, 0, 0, 0, 0, 0, 0, 0};
        __builtin_amdgcn_tensor_load_to_lds(g0, g1, z4, z4, z8, 0);
#else
        __builtin_amdgcn_tensor_load_to_lds(g0, g1, z4, z4, 0);
#endif
        __builtin_amdgcn_s_wait_tensorcnt(0);
    }
    __syncthreads();
#else
    {
        const uint4* src = (const uint4*)shn;               // 8192 x 16B
        uint4*       dst = (uint4*)lshn;
        for (int i = threadIdx.x; i < SHN_BYTES / 16; i += WAVES_PB * 32)
            dst[i] = src[i];
    }
    __syncthreads();
#endif

    // ---- tile assignment: 2 tiles per wave ----
    int wrow[TILES_PW];
    const float* invb = inv + (size_t)b * C_ * W_;
    const float* xb   = x   + (size_t)b * C_ * L_;
#pragma unroll
    for (int t = 0; t < TILES_PW; ++t) {
        int tl = blockIdx.x * TILES_PB + wv * TILES_PW + t;
        if (tl > WTILES - 1) tl = WTILES - 1;
        int w = tl * 16 + r16;
        wrow[t] = (w > W_ - 1) ? (W_ - 1) : w;
    }

    v8f acc[TILES_PW][4];
#pragma unroll
    for (int t = 0; t < TILES_PW; ++t)
#pragma unroll
        for (int kt = 0; kt < 4; ++kt) acc[t][kt] = (v8f){};

    for (int c = 0; c < C_; ++c) {
        // ---- preload all 8 B fragments for this channel in one go ----
        v16h bf[2][4];
#pragma unroll
        for (int sHalf = 0; sHalf < 2; ++sHalf) {
            const int soff = 32 * sHalf + 8 * halfId;
#pragma unroll
            for (int kt = 0; kt < 4; ++kt) {
                const _Float16* bp = lshn + ((c * K_ + kt * 16 + r16) * S_ + soff);
                ((uint4*)&bf[sHalf][kt])[0] = ((const uint4*)bp)[0];
                ((uint4*)&bf[sHalf][kt])[1] = ((const uint4*)bp)[2];
            }
        }

        const float* xc0 = xb + (size_t)c * L_ + wrow[0];
        const float* xc1 = xb + (size_t)c * L_ + wrow[1];
        const float  sc0 = invb[(size_t)c * W_ + wrow[0]];
        const float  sc1 = invb[(size_t)c * W_ + wrow[1]];

#pragma unroll
        for (int sHalf = 0; sHalf < 2; ++sHalf) {
            // ---- pack A fragments (per-lane row scale = window inv-norm)
            const int sbase = 32 * sHalf + 8 * halfId;
            v16h a0, a1;
#pragma unroll
            for (int i = 0; i < 8; ++i) {
                a0[i]     = (_Float16)(xc0[sbase + i]      * sc0);
                a0[i + 8] = (_Float16)(xc0[sbase + 16 + i] * sc0);
                a1[i]     = (_Float16)(xc1[sbase + i]      * sc1);
                a1[i + 8] = (_Float16)(xc1[sbase + 16 + i] * sc1);
            }
#pragma unroll
            for (int kt = 0; kt < 4; ++kt) {
                acc[0][kt] = __builtin_amdgcn_wmma_f32_16x16x32_f16(
                    false, a0, false, bf[sHalf][kt], (short)0, acc[0][kt], false, false);
                acc[1][kt] = __builtin_amdgcn_wmma_f32_16x16x32_f16(
                    false, a1, false, bf[sHalf][kt], (short)0, acc[1][kt], false, false);
            }
        }
    }

    // ---- relu, 1/C, reduce rows, atomic-max into out[b][k] ----
    float* outb = out + (size_t)b * K_;
#pragma unroll
    for (int t = 0; t < TILES_PW; ++t) {
#pragma unroll
        for (int kt = 0; kt < 4; ++kt) {
            float v = 0.0f;                      // relu
#pragma unroll
            for (int j = 0; j < 8; ++j) v = fmaxf(v, acc[t][kt][j]);
            v *= (1.0f / (float)C_);
            float o = __shfl_xor(v, 16, 32);     // merge M rows 0-7 / 8-15
            v = fmaxf(v, o);
            if (lane < 16)                       // v>=0: int cmp == float cmp
                atomicMax((int*)(outb + kt * 16 + r16), __float_as_int(v));
        }
    }
}

// ---------------------------------------------------------------- launcher
extern "C" void kernel_launch(void* const* d_in, const int* in_sizes, int n_in,
                              void* d_out, int out_size, void* d_ws, size_t ws_size,
                              hipStream_t stream) {
    const float* x  = (const float*)d_in[0];   // (32,16,4096) f32
    const float* sh = (const float*)d_in[1];   // (16,64,64)   f32
    float* out = (float*)d_out;                // (32,1,64)    f32

    _Float16* shn = (_Float16*)d_ws;                       // 128 KB
    float*    inv = (float*)((char*)d_ws + SHN_BYTES);     // 32*16*4033 f32

    zero_out_kernel<<<dim3((B_ * K_ + 255) / 256), 256, 0, stream>>>(out);
    shapelet_norm_kernel<<<dim3(4), 256, 0, stream>>>(sh, shn);
    window_invnorm_kernel<<<dim3((W_ + 255) / 256, B_ * C_), 256, 0, stream>>>(x, inv);

    const int gx = (WTILES + TILES_PB - 1) / TILES_PB;     // 32
    sim_wmma_kernel<<<dim3(gx, B_), WAVES_PB * 32, SHN_BYTES, stream>>>(
        x, shn, inv, out);
}
